// VCSWin3DBlock_30588757082821
// MI455X (gfx1250) — matrix operand order repaired and checked by reading, not compiled, not run on hardware
//
#include <hip/hip_runtime.h>
#include <hip/hip_bf16.h>
#include <math.h>

typedef _Float16 f16;
typedef __attribute__((ext_vector_type(16))) _Float16 v16h;
typedef __attribute__((ext_vector_type(8)))  float    v8f;

union U16H { v16h v; unsigned int u[8]; };
union U8F  { v8f  v; float        f[8]; };

__device__ __forceinline__ v8f v8f_zero() {
    U8F z;
#pragma unroll
    for (int i = 0; i < 8; ++i) z.f[i] = 0.0f;
    return z.v;
}

// ---- CDNA5 async global->LDS (ASYNCcnt-tracked DMA-style copy, 16B per lane) --
// LDS destination offset = low 32 bits of the generic pointer (LDS aperture rule).
__device__ __forceinline__ void async_copy16(const void* gsrc, void* ldst) {
    unsigned loff = (unsigned)(unsigned long long)ldst;
    unsigned long long g = (unsigned long long)gsrc;
    asm volatile("global_load_async_to_lds_b128 %0, %1, off"
                 :: "v"(loff), "v"(g) : "memory");
}
__device__ __forceinline__ void async_wait0() {
    asm volatile("s_wait_asynccnt 0x0" ::: "memory");
}

// A fragment: 16x32 f16, row-major tile in LDS with inner stride 32 halves.
// ISA layout: lane&15 = M row; pairs: VGPR j holds K = (j<4?0:16) + (laneHi?8:0) + 2*(j&3) .. +1
__device__ __forceinline__ v16h load_fragA(const f16* base) {
    int lane = threadIdx.x & 31;
    int m = lane & 15, hi = lane >> 4;
    U16H r;
#pragma unroll
    for (int j = 0; j < 8; ++j) {
        int k = ((j < 4) ? 0 : 16) + (hi ? 8 : 0) + 2 * (j & 3);
        r.u[j] = *(const unsigned int*)(base + m * 32 + k);
    }
    return r.v;
}

// B fragment: 32x16 f16. Tile stored transposed in LDS as [n][k], stride 32 halves.
// ISA layout: lane&15 = N col; pairs: VGPR j holds K = (j<4?0:16) + (laneHi?8:0) + 2*(j&3) .. +1
__device__ __forceinline__ v16h load_fragB(const f16* base) {
    int lane = threadIdx.x & 31;
    int n = lane & 15, hi = lane >> 4;
    U16H r;
#pragma unroll
    for (int j = 0; j < 8; ++j) {
        int k = ((j < 4) ? 0 : 16) + (hi ? 8 : 0) + 2 * (j & 3);
        r.u[j] = *(const unsigned int*)(base + n * 32 + k);
    }
    return r.v;
}

__device__ __forceinline__ v8f wmma16(v16h a, v16h b, v8f c) {
    return __builtin_amdgcn_wmma_f32_16x16x32_f16(false, a, false, b, (short)0, c, false, false);
}

// ---------------------------------------------------------------- cvt f32->f16
__global__ __launch_bounds__(256) void cvt_kernel(const float* __restrict__ s,
                                                  f16* __restrict__ d, int n) {
    int i = blockIdx.x * 256 + threadIdx.x;
    if (i < n) d[i] = (f16)s[i];
}

// ------------------------------------------------------------------- layernorm
// is_pre=1: src is x in (B,C,16,16,16); computes v=2*x, writes xs fp32 + LN(f16).
// is_pre=0: src is xs (row-major (B*L, C)); writes LN(f16) only.
__global__ __launch_bounds__(32) void ln_kernel(const float* __restrict__ src,
                                                const float* __restrict__ g,
                                                const float* __restrict__ bt,
                                                float* __restrict__ xsOut,
                                                f16* __restrict__ lnh, int is_pre) {
    int row = blockIdx.x;            // 0..8191
    int lane = threadIdx.x;          // wave32
    int b = row >> 12, l = row & 4095;
    float v[6];
#pragma unroll
    for (int j = 0; j < 6; ++j) {
        int c = lane + j * 32;
        v[j] = is_pre ? 2.0f * src[((size_t)(b * 192 + c)) * 4096 + l]
                      : src[(size_t)row * 192 + c];
    }
    float s = 0.f;
#pragma unroll
    for (int j = 0; j < 6; ++j) s += v[j];
#pragma unroll
    for (int m = 16; m >= 1; m >>= 1) s += __shfl_xor(s, m, 32);
    float mean = s * (1.0f / 192.0f);
    float q = 0.f;
#pragma unroll
    for (int j = 0; j < 6; ++j) { float d = v[j] - mean; q += d * d; }
#pragma unroll
    for (int m = 16; m >= 1; m >>= 1) q += __shfl_xor(q, m, 32);
    float rst = rsqrtf(q * (1.0f / 192.0f) + 1e-5f);
#pragma unroll
    for (int j = 0; j < 6; ++j) {
        int c = lane + j * 32;
        float y = (v[j] - mean) * rst * g[c] + bt[c];
        lnh[(size_t)row * 192 + c] = (f16)y;
        if (is_pre) xsOut[(size_t)row * 192 + c] = v[j];
    }
}

// -------------------------------------------------------------------- WMMA GEMM
// D = A(MxK,f16) @ Bw(KxN,f16) [+bias]; mode 0: store f16, 1: GELU->f16, 2: accF += D
__global__ __launch_bounds__(128) void gemm_kernel(const f16* __restrict__ A,
                                                   const f16* __restrict__ Bw,
                                                   const float* __restrict__ bias,
                                                   f16* __restrict__ outH,
                                                   float* __restrict__ accF,
                                                   int M, int N, int K, int mode) {
    __shared__ alignas(16) f16 Abuf[64 * 32];   // [m][k]
    __shared__ alignas(16) f16 Bbuf[64 * 32];   // [n][k] (transposed)
    int tid = threadIdx.x, wave = tid >> 5, lane = tid & 31;
    int row0 = blockIdx.y * 64, col0 = blockIdx.x * 64;
    U8F acc[4];
#pragma unroll
    for (int i = 0; i < 4; ++i) acc[i].v = v8f_zero();

    for (int kt = 0; kt < K / 32; ++kt) {
        __syncthreads();
        {   // stage A: 64 rows x 32 halves via async global->LDS (2x16B per lane)
            int r = tid >> 1, h = tid & 1;
            const f16* s = A + (size_t)(row0 + r) * K + kt * 32 + h * 16;
            f16* d = Abuf + r * 32 + h * 16;
            async_copy16(s, d);
            async_copy16(s + 8, d + 8);
        }
        // stage B transposed: Bbuf[n][k] = Bw[kt*32+k][col0+n]
#pragma unroll
        for (int j = 0; j < 16; ++j) {
            int idx = tid * 16 + j;
            int n = idx >> 5, k = idx & 31;
            Bbuf[n * 32 + k] = Bw[(size_t)(kt * 32 + k) * N + col0 + n];
        }
        async_wait0();
        __syncthreads();
        v16h af = load_fragA(Abuf + wave * 16 * 32);
#pragma unroll
        for (int nt = 0; nt < 4; ++nt) {
            v16h bf = load_fragB(Bbuf + nt * 16 * 32);
            acc[nt].v = wmma16(af, bf, acc[nt].v);
        }
    }
    int hi = lane >> 4, nl = lane & 15;
#pragma unroll
    for (int nt = 0; nt < 4; ++nt) {
        int gcol = col0 + nt * 16 + nl;
        float bv = bias ? bias[gcol] : 0.0f;   // hoisted: one load per col tile
#pragma unroll
        for (int r = 0; r < 8; ++r) {
            int grow = row0 + wave * 16 + r + 8 * hi;
            float v = acc[nt].f[r] + bv;
            if (mode == 0) {
                outH[(size_t)grow * N + gcol] = (f16)v;
            } else if (mode == 1) {
                float ge = 0.5f * v * (1.0f + erff(v * 0.70710678118654752f));
                outH[(size_t)grow * N + gcol] = (f16)ge;
            } else {
                accF[(size_t)grow * N + gcol] += v;
            }
        }
    }
}

// ----------------------------------------------------------- windowed attention
// qkv: (B*L, 576) f16 [q|k|v]; attnH: (B*L, 192) f16 output.
// Grid: ((branch*8 + b*4 + win)*2 + head)*16 + chunk  -> 768 blocks of 128 threads.
__global__ __launch_bounds__(128) void attn_kernel(const f16* __restrict__ qkv,
                                                   f16* __restrict__ attnH) {
    const int DSPt[3] = {16, 16, 4};
    const int HSPt[3] = {16, 4, 16};
    const int WSPt[3] = {4, 16, 16};

    int x = blockIdx.x;
    int chunk = x & 15;  x >>= 4;
    int head  = x & 1;   x >>= 1;
    int bw    = x & 7;   x >>= 3;
    int branch = x;
    int b = bw >> 2, w = bw & 3;

    int dsp = DSPt[branch], hsp = HSPt[branch], wsp = WSPt[branch];
    int nwh = 16 / hsp, nww = 16 / wsp;
    int wd = w / (nwh * nww), wh = (w / nww) % nwh, wwb = w % nww;

    auto tok2l = [&](int t) -> int {
        int dq = t / (hsp * wsp);
        int hq = (t / wsp) % hsp;
        int wq = t % wsp;
        int d = wd * dsp + dq, h2 = wh * hsp + hq, w2 = wwb * wsp + wq;
        return (d * 16 + h2) * 16 + w2;
    };

    int cbase = branch * 64 + head * 32;
    int tid = threadIdx.x, wave = tid >> 5, lane = tid & 31;
    int hi = lane >> 4, nl = lane & 15;
    const float scale = 0.17677669529663689f;   // 32^-0.5

    __shared__ alignas(16) f16 Qs[64 * 32];       // [qrow][ch]
    __shared__ alignas(16) f16 Ks[32 * 32];       // [tok][ch]
    __shared__ alignas(16) f16 Vs[32 * 32];       // [ch][tok] (transposed)
    __shared__ alignas(16) f16 Ps[4 * 16 * 32];   // per-wave [m][tok]

    // stage Q chunk (pre-scaled)
#pragma unroll
    for (int j = 0; j < 16; ++j) {
        int idx = tid * 16 + j;
        int m = idx >> 5, c = idx & 31;
        int l = tok2l(chunk * 64 + m);
        float qv = (float)qkv[((size_t)(b * 4096 + l)) * 576 + cbase + c];
        Qs[m * 32 + c] = (f16)(qv * scale);
    }
    __syncthreads();
    v16h qf = load_fragA(Qs + wave * 16 * 32);

    float mrow[8], lrow[8];
    U8F o0, o1;
    o0.v = v8f_zero(); o1.v = v8f_zero();
#pragma unroll
    for (int r = 0; r < 8; ++r) { mrow[r] = -3.0e38f; lrow[r] = 0.0f; }

    f16* pw = Ps + wave * 16 * 32;

    for (int kt = 0; kt < 32; ++kt) {
        __syncthreads();
        {   // stage K [tok][ch] via async DMA; V [ch][tok] via transpose scatter
            int t = tid >> 2, p = tid & 3;
            int l = tok2l(kt * 32 + t);
            size_t rb = ((size_t)(b * 4096 + l)) * 576;
            async_copy16(qkv + rb + 192 + cbase + p * 8, Ks + t * 32 + p * 8);
            uint4 vv = *(const uint4*)(qkv + rb + 384 + cbase + p * 8);
            const f16* vh = (const f16*)&vv;
#pragma unroll
            for (int i = 0; i < 8; ++i) Vs[(p * 8 + i) * 32 + t] = vh[i];
        }
        async_wait0();
        __syncthreads();

        v16h kf0 = load_fragB(Ks);
        v16h kf1 = load_fragB(Ks + 16 * 32);
        U8F s0, s1;
        s0.v = wmma16(qf, kf0, v8f_zero());
        s1.v = wmma16(qf, kf1, v8f_zero());

        // online softmax per row (rows live in 16-lane halves of the C layout)
#pragma unroll
        for (int r = 0; r < 8; ++r) {
            float a0 = s0.f[r], a1 = s1.f[r];
            float mx = fmaxf(a0, a1);
#pragma unroll
            for (int m = 8; m >= 1; m >>= 1) mx = fmaxf(mx, __shfl_xor(mx, m, 32));
            float mnew  = fmaxf(mrow[r], mx);
            float alpha = __expf(mrow[r] - mnew);
            float p0 = __expf(a0 - mnew), p1 = __expf(a1 - mnew);
            float rs = p0 + p1;
#pragma unroll
            for (int m = 8; m >= 1; m >>= 1) rs += __shfl_xor(rs, m, 32);
            lrow[r] = lrow[r] * alpha + rs;
            mrow[r] = mnew;
            o0.f[r] *= alpha;
            o1.f[r] *= alpha;
            int m = r + 8 * hi;
            pw[m * 32 + nl]      = (f16)p0;
            pw[m * 32 + 16 + nl] = (f16)p1;
        }
        // same-wave LDS RAW: drain DS counter (CDNA5 split counters)
        asm volatile("s_wait_dscnt 0" ::: "memory");

        v16h pf  = load_fragA(pw);            // A: 16 q-rows x 32 tokens
        v16h vf0 = load_fragB(Vs);            // B: 32 tokens x chans 0-15
        v16h vf1 = load_fragB(Vs + 16 * 32);  // B: 32 tokens x chans 16-31
        o0.v = wmma16(pf, vf0, o0.v);
        o1.v = wmma16(pf, vf1, o1.v);
    }

    // finalize and scatter back to (B,L,192)
#pragma unroll
    for (int r = 0; r < 8; ++r) {
        float inv = 1.0f / lrow[r];
        int m = r + 8 * hi;
        int l = tok2l(chunk * 64 + wave * 16 + m);
        size_t ob = ((size_t)(b * 4096 + l)) * 192 + cbase;
        attnH[ob + nl]      = (f16)(o0.f[r] * inv);
        attnH[ob + 16 + nl] = (f16)(o1.f[r] * inv);
    }
}

// --------------------------------------------------------------- final transpose
__global__ __launch_bounds__(256) void out_kernel(const float* __restrict__ xs,
                                                  float* __restrict__ out, int n) {
    int i = blockIdx.x * 256 + threadIdx.x;
    if (i >= n) return;
    int b = i / (192 * 4096);
    int c = (i / 4096) % 192;
    int l = i % 4096;
    out[i] = xs[((size_t)(b * 4096 + l)) * 192 + c];
}

// =============================================================================
extern "C" void kernel_launch(void* const* d_in, const int* in_sizes, int n_in,
                              void* d_out, int out_size, void* d_ws, size_t ws_size,
                              hipStream_t stream) {
    const float* x      = (const float*)d_in[0];
    const float* w_qkv  = (const float*)d_in[1];
    const float* w_proj = (const float*)d_in[2];
    const float* b_proj = (const float*)d_in[3];
    const float* g1     = (const float*)d_in[4];
    const float* b1     = (const float*)d_in[5];
    const float* g2     = (const float*)d_in[6];
    const float* b2     = (const float*)d_in[7];
    const float* w_fc1  = (const float*)d_in[8];
    const float* b_fc1  = (const float*)d_in[9];
    const float* w_fc2  = (const float*)d_in[10];
    const float* b_fc2  = (const float*)d_in[11];

    char* ws = (char*)d_ws;
    size_t off = 0;
    auto alloc = [&](size_t bytes) -> char* {
        char* p = ws + off;
        off = (off + bytes + 255) & ~(size_t)255;
        return p;
    };
    const int M = 8192;  // B*L
    f16*   wqkv_h  = (f16*)alloc(192 * 576 * 2);
    f16*   wproj_h = (f16*)alloc(192 * 192 * 2);
    f16*   wfc1_h  = (f16*)alloc(192 * 768 * 2);
    f16*   wfc2_h  = (f16*)alloc(768 * 192 * 2);
    float* xs      = (float*)alloc((size_t)M * 192 * 4);
    f16*   lnh     = (f16*)alloc((size_t)M * 192 * 2);
    f16*   qkvh    = (f16*)alloc((size_t)M * 576 * 2);
    f16*   attnh   = (f16*)alloc((size_t)M * 192 * 2);
    f16*   hidh    = (f16*)alloc((size_t)M * 768 * 2);

    // weights -> f16
    cvt_kernel<<<(192 * 576 + 255) / 256, 256, 0, stream>>>(w_qkv, wqkv_h, 192 * 576);
    cvt_kernel<<<(192 * 192 + 255) / 256, 256, 0, stream>>>(w_proj, wproj_h, 192 * 192);
    cvt_kernel<<<(192 * 768 + 255) / 256, 256, 0, stream>>>(w_fc1, wfc1_h, 192 * 768);
    cvt_kernel<<<(768 * 192 + 255) / 256, 256, 0, stream>>>(w_fc2, wfc2_h, 768 * 192);

    // 2x + transpose + LN1 -> xs (fp32 residual), lnh (f16)
    ln_kernel<<<M, 32, 0, stream>>>(x, g1, b1, xs, lnh, 1);

    // qkv = ln1 @ w_qkv : (8192 x 576)
    gemm_kernel<<<dim3(576 / 64, M / 64), 128, 0, stream>>>(
        lnh, wqkv_h, nullptr, qkvh, nullptr, M, 576, 192, 0);

    // 3-branch windowed flash attention
    attn_kernel<<<768, 128, 0, stream>>>(qkvh, attnh);

    // xs += attn @ w_proj + b_proj
    gemm_kernel<<<dim3(192 / 64, M / 64), 128, 0, stream>>>(
        attnh, wproj_h, b_proj, nullptr, xs, M, 192, 192, 2);

    // LN2 -> lnh (reuse)
    ln_kernel<<<M, 32, 0, stream>>>(xs, g2, b2, nullptr, lnh, 0);

    // hid = gelu(ln2 @ w_fc1 + b_fc1)
    gemm_kernel<<<dim3(768 / 64, M / 64), 128, 0, stream>>>(
        lnh, wfc1_h, b_fc1, hidh, nullptr, M, 768, 192, 1);

    // xs += hid @ w_fc2 + b_fc2
    gemm_kernel<<<dim3(192 / 64, M / 64), 128, 0, stream>>>(
        hidh, wfc2_h, b_fc2, nullptr, xs, M, 192, 768, 2);

    // (B,L,C) -> (B,C,D,H,W)
    out_kernel<<<(2 * 192 * 4096 + 255) / 256, 256, 0, stream>>>(
        xs, (float*)d_out, 2 * 192 * 4096);
}
